// Attn_4243427689113
// MI455X (gfx1250) — matrix operand order repaired and checked by reading, not compiled, not run on hardware
//
#include <hip/hip_runtime.h>

// ---------------------------------------------------------------------------
// MI455X (gfx1250, wave32) fused causal multi-head attention forward.
// All GEMMs use v_wmma_f32_16x16x32_bf16 (f32 accumulate).
// ---------------------------------------------------------------------------

typedef __bf16 bf16_t;
typedef __attribute__((ext_vector_type(16))) __bf16 v16bf;
typedef __attribute__((ext_vector_type(8)))  __bf16 bf16x8;
typedef __attribute__((ext_vector_type(8)))  float  v8f;

#define D_MODEL 1024
#define D_HEAD  64
#define N_HEADS 16
#define BSZ     2
#define N_CTX   2048
#define M_ROWS  (BSZ * N_CTX)   // 4096

union AFragU { v16bf v; bf16x8 h[2]; };

// A-matrix 16x32 bf16 layout (matching B): lane half hi: K = 16*hi+{0..15};
// this lane's row is contiguous in memory (stride-1 K).
__device__ __forceinline__ v16bf load_a_frag(const bf16_t* rowk0, int hi) {
  AFragU f;
  f.h[0] = *reinterpret_cast<const bf16x8*>(rowk0 + hi * 16);
  f.h[1] = *reinterpret_cast<const bf16x8*>(rowk0 + hi * 16 + 8);
  return f.v;
}

// B fragment, this code's contiguous K order: lane n holds column n of B; lane half hi holds
// K = 16*hi + {0..15}, 2 values per VGPR (k=2v,2v+1). With B[k][n] = W[n][k]
// (second operand stored row-major over K) this is one contiguous 32B load.
__device__ __forceinline__ v16bf load_b_frag(const bf16_t* colk0, int hi) {
  return *reinterpret_cast<const v16bf*>(colk0 + 16 * hi);
}

__device__ __forceinline__ v8f wmma_bf16(v16bf a, v16bf b, v8f c) {
  return __builtin_amdgcn_wmma_f32_16x16x32_bf16(
      /*neg_a=*/false, a, /*neg_b=*/false, b,
      /*c_mod=*/(short)0, c, /*reuse_a=*/false, /*reuse_b=*/false);
}

__device__ __forceinline__ v8f zero8() {
  v8f z;
#pragma unroll
  for (int i = 0; i < 8; ++i) z[i] = 0.0f;
  return z;
}

// ---------------------------------------------------------------------------
// fp32 -> bf16 conversion
// ---------------------------------------------------------------------------
__global__ void cvt_f32_bf16_kernel(const float* __restrict__ in,
                                    bf16_t* __restrict__ out, int n) {
  int i = blockIdx.x * 256 + threadIdx.x;
  if (i < n) out[i] = (bf16_t)in[i];
}

// ---------------------------------------------------------------------------
// GEMM: C[M][N] = (A[M][K] @ W[N][K]^T + bias) * scale
// mode 0: store bf16 head-major  out[((b*16+h)*2048+n)*64+d]   (QKV path)
// mode 1: store f32 row-major    out[m*N+c]                    (final proj)
// Block: 256 threads = 8 waves arranged 2(M) x 4(N); wave tile 64x64;
// block tile 128x256. K stepped by 32 (one WMMA depth).
// Per K-step per wave: 16 b128-equivalent loads feeding 16 WMMAs.
// ---------------------------------------------------------------------------
__global__ __launch_bounds__(256) void gemm_bf16_kernel(
    const bf16_t* __restrict__ A, const bf16_t* __restrict__ W,
    const float* __restrict__ bias, int M, int N, int K, int mode, float scale,
    bf16_t* __restrict__ outb, float* __restrict__ outf) {
  const int lane = threadIdx.x & 31;
  const int wave = threadIdx.x >> 5;
  const int hi = lane >> 4;
  const int lr = lane & 15;
  const int waveM = wave >> 2;  // 0..1
  const int waveN = wave & 3;   // 0..3
  const int rowBase = blockIdx.x * 128 + waveM * 64;
  const int colBase = blockIdx.y * 256 + waveN * 64;

  const bf16_t* Arow[4];
  const bf16_t* Wrow[4];
#pragma unroll
  for (int i = 0; i < 4; ++i)
    Arow[i] = A + (size_t)(rowBase + i * 16 + lr) * K;
#pragma unroll
  for (int j = 0; j < 4; ++j)
    Wrow[j] = W + (size_t)(colBase + j * 16 + lr) * K;

  v8f acc[4][4];
#pragma unroll
  for (int i = 0; i < 4; ++i)
#pragma unroll
    for (int j = 0; j < 4; ++j) acc[i][j] = zero8();

  for (int k0 = 0; k0 < K; k0 += 32) {
    // speculative prefetch a few K-steps ahead (safe: dropped on fault)
    __builtin_prefetch(Arow[0] + k0 + 256, 0, 3);
    __builtin_prefetch(Wrow[0] + k0 + 256, 0, 3);

    v16bf a[4], b[4];
#pragma unroll
    for (int i = 0; i < 4; ++i) a[i] = load_a_frag(Arow[i] + k0, hi);
#pragma unroll
    for (int j = 0; j < 4; ++j) b[j] = load_b_frag(Wrow[j] + k0, hi);
#pragma unroll
    for (int i = 0; i < 4; ++i)
#pragma unroll
      for (int j = 0; j < 4; ++j) acc[i][j] = wmma_bf16(a[i], b[j], acc[i][j]);
  }

  // hoist bias (one value per N-tile column owned by this lane)
  float bcol[4];
#pragma unroll
  for (int j = 0; j < 4; ++j)
    bcol[j] = bias ? bias[colBase + j * 16 + lr] : 0.0f;

  // C/D layout: lane half hi, VGPR v -> (m = v + 8*hi, n = lane&15)
#pragma unroll
  for (int i = 0; i < 4; ++i) {
#pragma unroll
    for (int j = 0; j < 4; ++j) {
#pragma unroll
      for (int v = 0; v < 8; ++v) {
        int m = rowBase + i * 16 + v + 8 * hi;
        int c = colBase + j * 16 + lr;
        float val = (acc[i][j][v] + bcol[j]) * scale;
        if (mode == 0) {
          int bb = m >> 11, n = m & (N_CTX - 1);
          int hh = c >> 6, d = c & (D_HEAD - 1);
          outb[(((size_t)bb * N_HEADS + hh) * N_CTX + n) * D_HEAD + d] =
              (bf16_t)val;
        } else {
          outf[(size_t)m * N + c] = val;
        }
      }
    }
  }
}

// ---------------------------------------------------------------------------
// Flash attention (causal). Grid: (BSZ*N_HEADS, N_CTX/128), 256 threads.
// Each wave owns 16 query rows (Q frags resident). Key blocks of 32.
// Q is pre-scaled by 1/sqrt(D_HEAD) at the QKV epilogue.
// ---------------------------------------------------------------------------
__global__ __launch_bounds__(256) void attn_kernel(
    const bf16_t* __restrict__ Qh, const bf16_t* __restrict__ Kh,
    const bf16_t* __restrict__ Vh, bf16_t* __restrict__ Oh) {
  __shared__ bf16_t VT[D_HEAD][48];   // V^T tile, padded rows (96B, 32B-aligned)
  __shared__ bf16_t Psh[8][16][48];   // per-wave P staging

  const int bh = blockIdx.x;          // b*16 + h
  const int b = bh >> 4, h = bh & 15;
  const int q0 = blockIdx.y * 128;

  const int tid = threadIdx.x;
  const int lane = tid & 31, wave = tid >> 5;
  const int hi = lane >> 4, lr = lane & 15;

  const bf16_t* Qp = Qh + (size_t)bh * N_CTX * D_HEAD;
  const bf16_t* Kp = Kh + (size_t)bh * N_CTX * D_HEAD;
  const bf16_t* Vp = Vh + (size_t)bh * N_CTX * D_HEAD;

  const int qrow = q0 + wave * 16 + lr;
  v16bf qf[2];
  qf[0] = load_a_frag(Qp + (size_t)qrow * D_HEAD + 0, hi);
  qf[1] = load_a_frag(Qp + (size_t)qrow * D_HEAD + 32, hi);

  v8f o[4];
#pragma unroll
  for (int t = 0; t < 4; ++t) o[t] = zero8();
  float rmax[8], rsum[8];
#pragma unroll
  for (int v = 0; v < 8; ++v) { rmax[v] = -1e30f; rsum[v] = 0.0f; }

  const int kend = q0 + 128;  // causal upper bound for this query block
  for (int kb = 0; kb < kend; kb += 32) {
    // prefetch next key block (speculative; tail overrun is harmless)
    __builtin_prefetch(Kp + (size_t)(kb + 32 + lane) * D_HEAD, 0, 3);
    __builtin_prefetch(Vp + (size_t)(kb + 32 + lane) * D_HEAD, 0, 3);

    // ---- cooperatively stage V^T (32 keys x 64 d -> VT[d][k]) ----
    __syncthreads();  // previous iteration's VT reads are done
    {
      int k = tid >> 3;            // 0..31
      int d0 = (tid & 7) * 8;      // 0,8,..,56
      bf16x8 vrow =
          *reinterpret_cast<const bf16x8*>(Vp + (size_t)(kb + k) * D_HEAD + d0);
#pragma unroll
      for (int e = 0; e < 8; ++e) VT[d0 + e][k] = vrow[e];
    }
    __syncthreads();

    // ---- S = Q K^T : two 16x16 tiles (keys kb..kb+15, kb+16..kb+31) ----
    v8f s0 = zero8(), s1 = zero8();
    {
      const bf16_t* kr0 = Kp + (size_t)(kb + lr) * D_HEAD;
      const bf16_t* kr1 = Kp + (size_t)(kb + 16 + lr) * D_HEAD;
      s0 = wmma_bf16(qf[0], load_b_frag(kr0 + 0, hi), s0);
      s0 = wmma_bf16(qf[1], load_b_frag(kr0 + 32, hi), s0);
      s1 = wmma_bf16(qf[0], load_b_frag(kr1 + 0, hi), s1);
      s1 = wmma_bf16(qf[1], load_b_frag(kr1 + 32, hi), s1);
    }

    // ---- causal mask + online softmax (row = VGPR v across 16 lanes) ----
    float p0[8], p1[8];
#pragma unroll
    for (int v = 0; v < 8; ++v) {
      int qi = q0 + wave * 16 + v + 8 * hi;
      float a = s0[v]; if (kb + lr > qi)      a = -1e38f;
      float c = s1[v]; if (kb + 16 + lr > qi) c = -1e38f;
      p0[v] = a; p1[v] = c;
    }
#pragma unroll
    for (int v = 0; v < 8; ++v) {
      float mx = fmaxf(p0[v], p1[v]);
      mx = fmaxf(mx, __shfl_xor(mx, 1, 32));
      mx = fmaxf(mx, __shfl_xor(mx, 2, 32));
      mx = fmaxf(mx, __shfl_xor(mx, 4, 32));
      mx = fmaxf(mx, __shfl_xor(mx, 8, 32));
      float newm = fmaxf(rmax[v], mx);
      float corr = __expf(rmax[v] - newm);
      rmax[v] = newm;
      float e0 = __expf(p0[v] - newm);
      float e1 = __expf(p1[v] - newm);
      p0[v] = e0; p1[v] = e1;
      float rs = e0 + e1;
      rs += __shfl_xor(rs, 1, 32);
      rs += __shfl_xor(rs, 2, 32);
      rs += __shfl_xor(rs, 4, 32);
      rs += __shfl_xor(rs, 8, 32);
      rsum[v] = rsum[v] * corr + rs;
#pragma unroll
      for (int t = 0; t < 4; ++t) o[t][v] *= corr;
    }

    // ---- P (16x32) C-layout -> LDS -> A-fragment ----
#pragma unroll
    for (int v = 0; v < 8; ++v) {
      int m = v + 8 * hi;
      Psh[wave][m][lr] = (bf16_t)p0[v];
      Psh[wave][m][16 + lr] = (bf16_t)p1[v];
    }
    // LDS ops from one wave are processed in order; keep compiler from
    // hoisting the reads above the writes and wait for DS completion.
    asm volatile("s_wait_dscnt 0" ::: "memory");
    v16bf pf = load_a_frag(&Psh[wave][lr][0], hi);

    // ---- O += P @ V (4 N-tiles of 16 along d) ----
#pragma unroll
    for (int t = 0; t < 4; ++t) {
      v16bf vf = load_b_frag(&VT[t * 16 + lr][0], hi);
      o[t] = wmma_bf16(pf, vf, o[t]);
    }
  }

  // ---- normalize and store O as bf16 in [b][n][h*64+d] ----
#pragma unroll
  for (int t = 0; t < 4; ++t) {
#pragma unroll
    for (int v = 0; v < 8; ++v) {
      int qi = q0 + wave * 16 + v + 8 * hi;
      int d = t * 16 + lr;
      Oh[((size_t)b * N_CTX + qi) * D_MODEL + h * D_HEAD + d] =
          (bf16_t)(o[t][v] / rsum[v]);
    }
  }
}

// ---------------------------------------------------------------------------
// Launcher
// ---------------------------------------------------------------------------
extern "C" void kernel_launch(void* const* d_in, const int* in_sizes, int n_in,
                              void* d_out, int out_size, void* d_ws,
                              size_t ws_size, hipStream_t stream) {
  const float* x  = (const float*)d_in[0];
  const float* Wq = (const float*)d_in[1];
  const float* bq = (const float*)d_in[2];
  const float* Wk = (const float*)d_in[3];
  const float* bk = (const float*)d_in[4];
  const float* Wv = (const float*)d_in[5];
  const float* bv = (const float*)d_in[6];
  const float* Wo = (const float*)d_in[7];
  float* out = (float*)d_out;

  // workspace layout (bf16 elements), ~48 MiB total
  bf16_t* ws = (bf16_t*)d_ws;
  size_t off = 0;
  bf16_t* xb  = ws + off; off += (size_t)M_ROWS * D_MODEL;     // 4096x1024
  bf16_t* wqb = ws + off; off += (size_t)D_MODEL * D_MODEL;
  bf16_t* wkb = ws + off; off += (size_t)D_MODEL * D_MODEL;
  bf16_t* wvb = ws + off; off += (size_t)D_MODEL * D_MODEL;
  bf16_t* wob = ws + off; off += (size_t)D_MODEL * D_MODEL;
  bf16_t* Qh  = ws + off; off += (size_t)M_ROWS * D_MODEL;     // [32][2048][64]
  bf16_t* Kh  = ws + off; off += (size_t)M_ROWS * D_MODEL;
  bf16_t* Vh  = ws + off; off += (size_t)M_ROWS * D_MODEL;
  bf16_t* Oh  = ws + off; off += (size_t)M_ROWS * D_MODEL;     // [4096][1024]

  const int nx = M_ROWS * D_MODEL;       // 4194304
  const int nw = D_MODEL * D_MODEL;      // 1048576
  cvt_f32_bf16_kernel<<<nx / 256, 256, 0, stream>>>(x,  xb,  nx);
  cvt_f32_bf16_kernel<<<nw / 256, 256, 0, stream>>>(Wq, wqb, nw);
  cvt_f32_bf16_kernel<<<nw / 256, 256, 0, stream>>>(Wk, wkb, nw);
  cvt_f32_bf16_kernel<<<nw / 256, 256, 0, stream>>>(Wv, wvb, nw);
  cvt_f32_bf16_kernel<<<nw / 256, 256, 0, stream>>>(Wo, wob, nw);

  dim3 ggrid(M_ROWS / 128, D_MODEL / 256);  // 32 x 4
  // Q projection gets the 1/sqrt(d_k) scale folded in.
  gemm_bf16_kernel<<<ggrid, 256, 0, stream>>>(xb, wqb, bq, M_ROWS, D_MODEL,
                                              D_MODEL, 0, 0.125f, Qh, nullptr);
  gemm_bf16_kernel<<<ggrid, 256, 0, stream>>>(xb, wkb, bk, M_ROWS, D_MODEL,
                                              D_MODEL, 0, 1.0f, Kh, nullptr);
  gemm_bf16_kernel<<<ggrid, 256, 0, stream>>>(xb, wvb, bv, M_ROWS, D_MODEL,
                                              D_MODEL, 0, 1.0f, Vh, nullptr);

  dim3 agrid(BSZ * N_HEADS, N_CTX / 128);   // 32 x 16
  attn_kernel<<<agrid, 256, 0, stream>>>(Qh, Kh, Vh, Oh);

  gemm_bf16_kernel<<<ggrid, 256, 0, stream>>>(Oh, wob, nullptr, M_ROWS,
                                              D_MODEL, D_MODEL, 1, 1.0f,
                                              nullptr, out);
}
